// DecSwitchedFC_44985487458667
// MI455X (gfx1250) — matrix operand order repaired and checked by reading, not compiled, not run on hardware
//
#include <hip/hip_runtime.h>

typedef __attribute__((ext_vector_type(16))) _Float16 v16h;
typedef __attribute__((ext_vector_type(8)))  _Float16 v8h;
typedef __attribute__((ext_vector_type(8)))  float    v8f;

namespace {
constexpr int Bn = 32768;   // batch rows
constexpr int Dd = 512;     // model dim
constexpr int Ss = 128;     // hidden dim
constexpr int En = 8;       // experts
constexpr int BM = 32;      // rows per block (2 WMMA row-tiles)
constexpr int AP = Dd + 8;  // LDS pitch (halfs) for A tile
constexpr int HP = Ss + 8;  // LDS pitch (halfs) for H tile
constexpr int NTHREADS = 256;                 // 8 waves (wave32)
constexpr int MAX_TILES = Bn / BM + En;       // 1032, covers worst-case padding

// workspace layout (bytes)
constexpr size_t CNT_OFF  = 0;                                        // 8 int
constexpr size_t CUR_OFF  = 64;                                       // 8 int
constexpr size_t AOFF_OFF = 128;                                      // 9 int
constexpr size_t RIDX_OFF = 256;                                      // (Bn + En*BM) int
constexpr size_t W1H_OFF  = RIDX_OFF + (size_t)(Bn + En * BM) * 4;    // E*S*D halfs (W1^T)
constexpr size_t W2H_OFF  = W1H_OFF + (size_t)En * Ss * Dd * 2;       // E*D*S halfs (W2^T)
}

// ---------------- routing / bucketing ----------------

__global__ void k_zero(int* cnt, int* cur) {
  if (threadIdx.x < En) { cnt[threadIdx.x] = 0; cur[threadIdx.x] = 0; }
}

__global__ void k_count(const int* __restrict__ yi, int* __restrict__ cnt) {
  int r = blockIdx.x * blockDim.x + threadIdx.x;
  if (r < Bn) atomicAdd(&cnt[yi[r]], 1);
}

__global__ void k_offsets(const int* __restrict__ cnt, int* __restrict__ aoff,
                          int* __restrict__ cur) {
  if (threadIdx.x == 0) {
    int a = 0;
    for (int e = 0; e < En; ++e) {
      aoff[e] = a; cur[e] = a;
      a += ((cnt[e] + BM - 1) / BM) * BM;   // 32-aligned segments
    }
    aoff[En] = a;
  }
}

__global__ void k_scatter(const int* __restrict__ yi, int* __restrict__ cur,
                          int* __restrict__ ridx) {
  int r = blockIdx.x * blockDim.x + threadIdx.x;
  if (r < Bn) {
    int pos = atomicAdd(&cur[yi[r]], 1);
    ridx[pos] = r;
  }
}

// ---------------- weight transpose + f32->f16 ----------------
// W1h[e][s][d] = W1[e][d][s]   (B-columns of GEMM1 become contiguous in K)
// W2h[e][d][s] = W2[e][s][d]   (B-columns of GEMM2 become contiguous in K)
__global__ void k_convert(const float* __restrict__ W1, const float* __restrict__ W2,
                          _Float16* __restrict__ W1h, _Float16* __restrict__ W2h) {
  int idx = blockIdx.x * blockDim.x + threadIdx.x;   // [0, En*Ss*Dd)
  {
    int e = idx / (Ss * Dd);
    int r = idx % (Ss * Dd);
    int s = r / Dd;
    int d = r % Dd;
    W1h[idx] = (_Float16)W1[((size_t)e * Dd + d) * Ss + s];
  }
  {
    int e = idx / (Dd * Ss);
    int r = idx % (Dd * Ss);
    int d = r / Ss;
    int s = r % Ss;
    W2h[idx] = (_Float16)W2[((size_t)e * Ss + s) * Dd + d];
  }
}

// ---------------- main grouped-GEMM MoE kernel ----------------

__global__ __launch_bounds__(NTHREADS) void k_moe(
    const float* __restrict__ x, const float* __restrict__ z,
    const float* __restrict__ b1, const float* __restrict__ b2,
    const _Float16* __restrict__ W1h, const _Float16* __restrict__ W2h,
    const int* __restrict__ cnt, const int* __restrict__ aoff,
    const int* __restrict__ ridx, float* __restrict__ out)
{
  __shared__ _Float16 Ah[BM * AP];   // relu(x) tile, f16
  __shared__ _Float16 Hh[BM * HP];   // hidden tile, f16
  __shared__ int   s_rid[BM];
  __shared__ float s_z[BM];

  const int start = blockIdx.x * BM;           // position in packed-aligned row array
  if (start >= aoff[En]) return;               // uniform exit
  int e = 0;
  while (e + 1 < En && aoff[e + 1] <= start) ++e;
  int nvalid = cnt[e] - (start - aoff[e]);
  if (nvalid <= 0) return;
  if (nvalid > BM) nvalid = BM;

  const int tid = threadIdx.x;
  if (tid < BM) {
    int rid = (tid < nvalid) ? ridx[start + tid] : -1;
    s_rid[tid] = rid;
    s_z[tid] = (rid >= 0) ? z[rid] : 0.0f;
  }
  __syncthreads();

  // gather x rows, relu, convert to f16 into LDS (coalesced within a row)
  for (int i = tid; i < BM * Dd; i += NTHREADS) {
    int row = i >> 9;            // / Dd
    int col = i & (Dd - 1);
    int rid = s_rid[row];
    float v = (rid >= 0) ? x[(size_t)rid * Dd + col] : 0.0f;
    v = v > 0.0f ? v : 0.0f;
    Ah[row * AP + col] = (_Float16)v;
  }
  __syncthreads();

  const int wave = tid >> 5;
  const int lane = tid & 31;
  const int lm   = lane & 15;    // M (A/C) or N (B/C) index within tile
  const int lsel = lane >> 4;    // half-wave select

  union V16 { v16h v; v8h h[2]; };

  // ---- GEMM1: H = relu(A(32x512) @ W1[e](512x128) + b1[e]); 1 col-tile/wave
  {
    const int n0 = wave * 16;
    const float b1v = b1[e * Ss + n0 + lm];
    const _Float16* wp = W1h + (size_t)e * Ss * Dd + (size_t)(n0 + lm) * Dd + lsel * 16;
    v8f acc0 = {}; v8f acc1 = {};
#pragma unroll
    for (int ks = 0; ks < Dd / 32; ++ks) {
      const int k0 = ks * 32;
      v16h bf = *(const v16h*)(wp + k0);                 // contiguous 32B: K-run for col n0+lm
      V16 a0, a1;
      const _Float16* ap0 = &Ah[lm * AP + k0 + lsel * 16];
      a0.h[0] = *(const v8h*)(ap0);
      a0.h[1] = *(const v8h*)(ap0 + 8);
      const _Float16* ap1 = &Ah[(16 + lm) * AP + k0 + lsel * 16];
      a1.h[0] = *(const v8h*)(ap1);
      a1.h[1] = *(const v8h*)(ap1 + 8);
      acc0 = __builtin_amdgcn_wmma_f32_16x16x32_f16(false, a0.v, false, bf, (short)0, acc0, false, false);
      acc1 = __builtin_amdgcn_wmma_f32_16x16x32_f16(false, a1.v, false, bf, (short)0, acc1, false, false);
    }
#pragma unroll
    for (int v = 0; v < 8; ++v) {
      int m = lsel ? (v + 8) : v;                        // C layout: lanes16-31 hold M+8
      float h0 = acc0[v] + b1v; h0 = h0 > 0.f ? h0 : 0.f;
      float h1 = acc1[v] + b1v; h1 = h1 > 0.f ? h1 : 0.f;
      Hh[m * HP + n0 + lm]        = (_Float16)h0;
      Hh[(16 + m) * HP + n0 + lm] = (_Float16)h1;
    }
  }
  __syncthreads();

  // ---- GEMM2: O = H(32x128) @ W2[e](128x512) + b2[e]; out = x + O*z ; 4 col-tiles/wave
#pragma unroll
  for (int j = 0; j < 4; ++j) {
    const int n0 = (wave * 4 + j) * 16;
    const int dcol = n0 + lm;
    const float b2v = b2[e * Dd + dcol];
    const _Float16* wp = W2h + (size_t)e * Dd * Ss + (size_t)dcol * Ss + lsel * 16;
    v8f acc0 = {}; v8f acc1 = {};
#pragma unroll
    for (int ks = 0; ks < Ss / 32; ++ks) {
      const int k0 = ks * 32;
      v16h bf = *(const v16h*)(wp + k0);
      V16 a0, a1;
      const _Float16* ap0 = &Hh[lm * HP + k0 + lsel * 16];
      a0.h[0] = *(const v8h*)(ap0);
      a0.h[1] = *(const v8h*)(ap0 + 8);
      const _Float16* ap1 = &Hh[(16 + lm) * HP + k0 + lsel * 16];
      a1.h[0] = *(const v8h*)(ap1);
      a1.h[1] = *(const v8h*)(ap1 + 8);
      acc0 = __builtin_amdgcn_wmma_f32_16x16x32_f16(false, a0.v, false, bf, (short)0, acc0, false, false);
      acc1 = __builtin_amdgcn_wmma_f32_16x16x32_f16(false, a1.v, false, bf, (short)0, acc1, false, false);
    }
#pragma unroll
    for (int v = 0; v < 8; ++v) {
      int m = lsel ? (v + 8) : v;
      int r0 = m, r1 = 16 + m;
      int rid0 = s_rid[r0];
      if (rid0 >= 0) {
        size_t o = (size_t)rid0 * Dd + dcol;
        out[o] = x[o] + (acc0[v] + b2v) * s_z[r0];       // x re-read hits L2
      }
      int rid1 = s_rid[r1];
      if (rid1 >= 0) {
        size_t o = (size_t)rid1 * Dd + dcol;
        out[o] = x[o] + (acc1[v] + b2v) * s_z[r1];
      }
    }
  }
}

// ---------------- launcher ----------------

extern "C" void kernel_launch(void* const* d_in, const int* in_sizes, int n_in,
                              void* d_out, int out_size, void* d_ws, size_t ws_size,
                              hipStream_t stream) {
  const float* x  = (const float*)d_in[0];
  const int*   yi = (const int*)d_in[1];
  // d_in[2] (y_hard) is unused by the reference
  const float* z  = (const float*)d_in[3];
  const float* W1 = (const float*)d_in[4];
  const float* b1 = (const float*)d_in[5];
  const float* W2 = (const float*)d_in[6];
  const float* b2 = (const float*)d_in[7];
  float* out = (float*)d_out;

  char* ws = (char*)d_ws;
  int* cnt  = (int*)(ws + CNT_OFF);
  int* cur  = (int*)(ws + CUR_OFF);
  int* aoff = (int*)(ws + AOFF_OFF);
  int* ridx = (int*)(ws + RIDX_OFF);
  _Float16* W1h = (_Float16*)(ws + W1H_OFF);
  _Float16* W2h = (_Float16*)(ws + W2H_OFF);

  k_zero   <<<1, 32, 0, stream>>>(cnt, cur);
  k_count  <<<Bn / 256, 256, 0, stream>>>(yi, cnt);
  k_offsets<<<1, 1, 0, stream>>>(cnt, aoff, cur);
  k_scatter<<<Bn / 256, 256, 0, stream>>>(yi, cur, ridx);
  k_convert<<<(En * Ss * Dd) / 256, 256, 0, stream>>>(W1, W2, W1h, W2h);
  k_moe    <<<MAX_TILES, NTHREADS, 0, stream>>>(x, z, b1, b2, W1h, W2h, cnt, aoff, ridx, out);
}
